// RGCN_67319317398020
// MI455X (gfx1250) — matrix operand order repaired and checked by reading, not compiled, not run on hardware
//
#include <hip/hip_runtime.h>
#include <hip/hip_bf16.h>

typedef __attribute__((ext_vector_type(16))) __bf16        v16bf;
typedef __attribute__((ext_vector_type(8)))  float         v8f;
typedef __attribute__((ext_vector_type(8)))  unsigned int  v8u;

constexpr int R  = 4;
constexpr int FD = 128;
constexpr int FP = 256;
constexpr int FO = 128;

static __device__ __forceinline__ unsigned pack2_bf16(float a, float b) {
  unsigned ua = __float_as_uint(a);
  unsigned ub = __float_as_uint(b);
  ua = (ua + 0x7FFFu + ((ua >> 16) & 1u)) >> 16;   // round-to-nearest-even
  ub = (ub + 0x7FFFu + ((ub >> 16) & 1u)) >> 16;
  return (ua & 0xFFFFu) | (ub << 16);
}

__global__ void zero_f32_kernel(float* __restrict__ p, int n) {
  int i = blockIdx.x * blockDim.x + threadIdx.x;
  if (i < n) p[i] = 0.0f;
}

__global__ void zero_i32_kernel(int* __restrict__ p, int n) {
  int i = blockIdx.x * blockDim.x + threadIdx.x;
  if (i < n) p[i] = 0;
}

// Per-edge degree histogram: out-degree of src, in-degree count of dst (CSR row sizes).
__global__ void count_kernel(const int* __restrict__ src, const int* __restrict__ dst,
                             int* __restrict__ doutC, int* __restrict__ cnt, int E) {
  int e = blockIdx.x * blockDim.x + threadIdx.x;
  if (e < E) {
    atomicAdd(&doutC[src[e]], 1);
    atomicAdd(&cnt[dst[e]], 1);
  }
}

// Single-block exclusive scan of cnt[0..n) -> rowStart[0..n], cursor copy for fill.
__global__ void __launch_bounds__(1024)
scan_kernel(const int* __restrict__ cnt, int* __restrict__ rowStart,
            int* __restrict__ cursor, int n) {
  __shared__ int part[1024];
  const int t   = threadIdx.x;
  const int per = (n + 1023) / 1024;
  const int beg = t * per;
  const int end = (beg + per < n) ? (beg + per) : n;
  int sum = 0;
  for (int i = beg; i < end; ++i) sum += cnt[i];
  part[t] = sum;
  __syncthreads();
  for (int off = 1; off < 1024; off <<= 1) {      // Hillis-Steele inclusive scan
    int tmp = (t >= off) ? part[t - off] : 0;
    __syncthreads();
    part[t] += tmp;
    __syncthreads();
  }
  int run = part[t] - sum;                        // exclusive prefix of this chunk
  for (int i = beg; i < end; ++i) {
    rowStart[i] = run;
    cursor[i]   = run;
    run += cnt[i];
  }
  if (end == n) rowStart[n] = run;                // total (benign multi-writer, same value)
}

// CSR fill: bucket source indices by destination row.
__global__ void fill_kernel(const int* __restrict__ src, const int* __restrict__ dst,
                            int* __restrict__ cursor, int* __restrict__ eSrc, int E) {
  int e = blockIdx.x * blockDim.x + threadIdx.x;
  if (e < E) {
    int slot = atomicAdd(&cursor[dst[e]], 1);
    eSrc[slot] = src[e];
  }
}

// Pack X (rows x K f32) into bf16 A-fragments: Xpk[((tile*nChunks+kc)*32+lane)*8+v].
// A 16x32 bf16 layout: lane = half*16+q -> M=q; V0-3: K=half*8+v*2; V4-7: +16.
__global__ void pack_x_kernel(const float* __restrict__ X, unsigned* __restrict__ Xpk,
                              int K, int total) {
  int i = blockIdx.x * blockDim.x + threadIdx.x;
  if (i >= total) return;
  const int v    = i & 7;
  const int lane = (i >> 3) & 31;
  const int nChunks = K >> 5;
  const int kc   = (i >> 8) % nChunks;
  const int t    = (i >> 8) / nChunks;
  const int half = lane >> 4;
  const int q    = lane & 15;
  const int row  = t * 16 + q;
  const int k    = kc * 32 + half * 8 + ((v < 4) ? v * 2 : 16 + (v - 4) * 2);
  const float2 xv = *(const float2*)(X + (size_t)row * K + k);
  Xpk[i] = pack2_bf16(xv.x, xv.y);
}

// Pack W (K x FO f32) into bf16 B-fragments: Wpk[((kc*8+nt)*32+lane)*8+v].
// B 32x16 bf16 layout: lane = half*16+q -> N=q; V0-3: K=half*8+v*2; V4-7: +16.
__global__ void pack_w_kernel(const float* __restrict__ W, unsigned* __restrict__ Wpk,
                              int total) {
  int i = blockIdx.x * blockDim.x + threadIdx.x;
  if (i >= total) return;
  const int v    = i & 7;
  const int lane = (i >> 3) & 31;
  const int nt   = (i >> 8) & 7;
  const int kc   = i >> 11;
  const int half = lane >> 4;
  const int q    = lane & 15;
  const int k    = kc * 32 + half * 8 + ((v < 4) ? v * 2 : 16 + (v - 4) * 2);
  const int n    = nt * 16 + q;
  Wpk[i] = pack2_bf16(W[(size_t)k * FO + n], W[(size_t)(k + 1) * FO + n]);
}

// H = Xpk @ Wpk (bf16 in, f32 out). One wave per 16-row strip, 8 C tiles (FO=128).
// Register double-buffer, last chunk peeled: the loop body prefetches
// unconditionally (no zero-init, no guard) while 8 WMMAs execute.
__global__ void __launch_bounds__(128)
wmma_gemm_kernel(const v8u* __restrict__ Apk, const v8u* __restrict__ Bpk,
                 float* __restrict__ H, int nChunks, int nTiles) {
  const int wave = threadIdx.x >> 5;
  const int lane = threadIdx.x & 31;
  const int tile = blockIdx.x * 4 + wave;
  if (tile >= nTiles) return;                 // wave-uniform: EXEC all-1s below
  const int half = lane >> 4;
  const int q    = lane & 15;

  v8f acc[8] = {};
  const v8u* aPtr = Apk + (size_t)tile * nChunks * 32 + lane;
  const v8u* bPtr = Bpk + lane;

  v8u aCur = aPtr[0];
  v8u bCur[8];
#pragma unroll
  for (int nt = 0; nt < 8; ++nt) bCur[nt] = bPtr[(size_t)nt * 32];

#pragma unroll 2
  for (int kc = 0; kc + 1 < nChunks; ++kc) {
    const v8u aNxt = aPtr[(size_t)(kc + 1) * 32];
    v8u bNxt[8];
#pragma unroll
    for (int nt = 0; nt < 8; ++nt)
      bNxt[nt] = bPtr[(((size_t)(kc + 1)) * 8 + nt) * 32];

    const v16bf a = __builtin_bit_cast(v16bf, aCur);
#pragma unroll
    for (int nt = 0; nt < 8; ++nt) {
      const v16bf b = __builtin_bit_cast(v16bf, bCur[nt]);
      acc[nt] = __builtin_amdgcn_wmma_f32_16x16x32_bf16(
          false, a, false, b, (short)0, acc[nt], false, false);
    }
    aCur = aNxt;
#pragma unroll
    for (int nt = 0; nt < 8; ++nt) bCur[nt] = bNxt[nt];
  }

  {                                           // final chunk: no prefetch
    const v16bf a = __builtin_bit_cast(v16bf, aCur);
#pragma unroll
    for (int nt = 0; nt < 8; ++nt) {
      const v16bf b = __builtin_bit_cast(v16bf, bCur[nt]);
      acc[nt] = __builtin_amdgcn_wmma_f32_16x16x32_bf16(
          false, a, false, b, (short)0, acc[nt], false, false);
    }
  }

  // D layout: VGPR p, lane = half*16 + q -> (M = p + half*8, N = q)
#pragma unroll
  for (int nt = 0; nt < 8; ++nt) {
#pragma unroll
    for (int p = 0; p < 8; ++p) {
      const int m = tile * 16 + p + half * 8;
      const int n = nt * 16 + q;
      H[(size_t)m * FO + n] = acc[nt][p];
    }
  }
}

// Atomic-free CSR gather: one wave per destination row.
// acc[d] += rsqrt(max(indeg,1)) * sum_j H[src_j] * rsqrt(max(dout[src_j],1))
__global__ void __launch_bounds__(256)
gather_kernel(const float* __restrict__ H, const int* __restrict__ eSrc,
              const int* __restrict__ rowStart, const int* __restrict__ doutC,
              float* __restrict__ acc, int nRows) {
  const int wave = threadIdx.x >> 5;
  const int lane = threadIdx.x & 31;
  const int d = blockIdx.x * 8 + wave;
  if (d >= nRows) return;
  const int beg = rowStart[d];
  const int end = rowStart[d + 1];
  float4 a = {0.0f, 0.0f, 0.0f, 0.0f};
  for (int j = beg; j < end; ++j) {
    const int   s  = eSrc[j];
    const float sc = rsqrtf(fmaxf((float)doutC[s], 1.0f));
    const float4 v = *(const float4*)(H + (size_t)s * FO + lane * 4);
    a.x += v.x * sc; a.y += v.y * sc; a.z += v.z * sc; a.w += v.w * sc;
  }
  const float di = rsqrtf(fmaxf((float)(end - beg), 1.0f));
  float* out = acc + (size_t)d * FO + lane * 4;
  float4 cur = *(const float4*)out;
  cur.x += a.x * di; cur.y += a.y * di; cur.z += a.z * di; cur.w += a.w * di;
  *(float4*)out = cur;
}

// out = (acc + sum_r bR[r] + bX?) with optional ReLU
__global__ void bias_act_kernel(const float* __restrict__ acc, const float* __restrict__ bR,
                                const float* __restrict__ bX, float* __restrict__ out,
                                int n, int relu) {
  int i = blockIdx.x * blockDim.x + threadIdx.x;
  if (i < n) {
    const int f = i & (FO - 1);
    float b = bR[f] + bR[FO + f] + bR[2 * FO + f] + bR[3 * FO + f];
    if (bX) b += bX[f];
    float v = acc[i] + b;
    out[i] = relu ? fmaxf(v, 0.0f) : v;
  }
}

extern "C" void kernel_launch(void* const* d_in, const int* in_sizes, int n_in,
                              void* d_out, int out_size, void* d_ws, size_t ws_size,
                              hipStream_t stream) {
  const float* x_d   = (const float*)d_in[0];
  const float* x_p   = (const float*)d_in[1];
  const float* W1    = (const float*)d_in[2];
  const float* b1    = (const float*)d_in[3];
  const float* W1pdi = (const float*)d_in[4];
  const float* b1pdi = (const float*)d_in[5];
  const float* W2    = (const float*)d_in[8];
  const float* b2    = (const float*)d_in[9];
  const float* W3    = (const float*)d_in[10];
  const float* b3    = (const float*)d_in[11];
  const int* ddi_src = (const int*)d_in[12];
  const int* ddi_dst = (const int*)d_in[13];
  const int* pdi_src = (const int*)d_in[14];
  const int* pdi_dst = (const int*)d_in[15];

  const int ND    = in_sizes[0] / FD;
  const int NP    = in_sizes[1] / FP;
  const int E_DDI = in_sizes[12] / R;
  const int E_PDI = in_sizes[14];
  const int NF    = ND * FO;

  // ---- workspace layout ----
  float* ws  = (float*)d_ws;
  float* acc = ws;                    // ND*FO
  float* hA  = acc + (size_t)NF;      // ND*FO (per-relation GEMM output)
  float* hB  = hA + (size_t)NF;       // ND*FO (layer output ping buffer)
  unsigned* Wpk = (unsigned*)(hB + (size_t)NF);      // FP*FO/2 uints (64 KB max)
  unsigned* Xpk = Wpk + (size_t)FP * FO / 2;         // max(ND*FD, NP*FP)/2 uints
  int* ip = (int*)(Xpk + (size_t)(ND * FD > NP * FP ? ND * FD : NP * FP) / 2);
  // CSR per relation: 0..3 = DDI relations, 4 = PDI (all keyed by drug dst rows)
  int* rowStart[5];
  int* eSrc[5];
  int* doutC[5];
  for (int rel = 0; rel < 5; ++rel) {
    rowStart[rel] = ip;             ip += ND + 1;
    eSrc[rel]     = ip;             ip += (rel < 4) ? E_DDI : E_PDI;
    doutC[rel]    = ip;             ip += (rel < 4) ? ND : NP;
  }
  int* cntTmp    = ip;              ip += ND;
  int* cursorTmp = ip;              ip += ND;

  const dim3 blk(256);
  auto cdiv = [](int a, int b) { return (a + b - 1) / b; };

  // ---- CSR build (once per call; reused by all layers) ----
  for (int rel = 0; rel < 5; ++rel) {
    const int* srcP = (rel < 4) ? ddi_src + (size_t)rel * E_DDI : pdi_src;
    const int* dstP = (rel < 4) ? ddi_dst + (size_t)rel * E_DDI : pdi_dst;
    const int  E    = (rel < 4) ? E_DDI : E_PDI;
    const int  nSrc = (rel < 4) ? ND : NP;
    zero_i32_kernel<<<cdiv(nSrc, 256), blk, 0, stream>>>(doutC[rel], nSrc);
    zero_i32_kernel<<<cdiv(ND, 256), blk, 0, stream>>>(cntTmp, ND);
    count_kernel<<<cdiv(E, 256), blk, 0, stream>>>(srcP, dstP, doutC[rel], cntTmp, E);
    scan_kernel<<<1, 1024, 0, stream>>>(cntTmp, rowStart[rel], cursorTmp, ND);
    fill_kernel<<<cdiv(E, 256), blk, 0, stream>>>(srcP, dstP, cursorTmp, eSrc[rel], E);
  }

  const int tilesD = ND / 16;
  const int tilesP = NP / 16;
  const int xpkD   = ND * FD / 2;   // uints
  const int xpkP   = NP * FP / 2;
  const int wpkD   = FD * FO / 2;
  const int wpkP   = FP * FO / 2;
  const int gBlks  = cdiv(ND, 8);   // gather: 8 waves (rows) per 256-thread block

  // ---------------- Layer 1: DDI(R) + PDI, ReLU ----------------
  zero_f32_kernel<<<cdiv(NF, 256), blk, 0, stream>>>(acc, NF);
  pack_x_kernel<<<cdiv(xpkD, 256), blk, 0, stream>>>(x_d, Xpk, FD, xpkD);
  for (int r = 0; r < R; ++r) {
    pack_w_kernel<<<cdiv(wpkD, 256), blk, 0, stream>>>(W1 + (size_t)r * FD * FO, Wpk, wpkD);
    wmma_gemm_kernel<<<cdiv(tilesD, 4), dim3(128), 0, stream>>>(
        (const v8u*)Xpk, (const v8u*)Wpk, hA, FD / 32, tilesD);
    gather_kernel<<<gBlks, blk, 0, stream>>>(hA, eSrc[r], rowStart[r], doutC[r], acc, ND);
  }
  pack_x_kernel<<<cdiv(xpkP, 256), blk, 0, stream>>>(x_p, Xpk, FP, xpkP);
  pack_w_kernel<<<cdiv(wpkP, 256), blk, 0, stream>>>(W1pdi, Wpk, wpkP);
  wmma_gemm_kernel<<<cdiv(tilesP, 4), dim3(128), 0, stream>>>(
      (const v8u*)Xpk, (const v8u*)Wpk, hA, FP / 32, tilesP);
  gather_kernel<<<gBlks, blk, 0, stream>>>(hA, eSrc[4], rowStart[4], doutC[4], acc, ND);
  bias_act_kernel<<<cdiv(NF, 256), blk, 0, stream>>>(acc, b1, b1pdi, hB, NF, 1);

  // ---------------- Layer 2: DDI(R), ReLU ----------------
  zero_f32_kernel<<<cdiv(NF, 256), blk, 0, stream>>>(acc, NF);
  pack_x_kernel<<<cdiv(xpkD, 256), blk, 0, stream>>>(hB, Xpk, FO, xpkD);
  for (int r = 0; r < R; ++r) {
    pack_w_kernel<<<cdiv(wpkD, 256), blk, 0, stream>>>(W2 + (size_t)r * FO * FO, Wpk, wpkD);
    wmma_gemm_kernel<<<cdiv(tilesD, 4), dim3(128), 0, stream>>>(
        (const v8u*)Xpk, (const v8u*)Wpk, hA, FO / 32, tilesD);
    gather_kernel<<<gBlks, blk, 0, stream>>>(hA, eSrc[r], rowStart[r], doutC[r], acc, ND);
  }
  bias_act_kernel<<<cdiv(NF, 256), blk, 0, stream>>>(acc, b2, nullptr, hB, NF, 1);

  // ---------------- Layer 3: DDI(R), no ReLU -> d_out ----------------
  zero_f32_kernel<<<cdiv(NF, 256), blk, 0, stream>>>(acc, NF);
  pack_x_kernel<<<cdiv(xpkD, 256), blk, 0, stream>>>(hB, Xpk, FO, xpkD);
  for (int r = 0; r < R; ++r) {
    pack_w_kernel<<<cdiv(wpkD, 256), blk, 0, stream>>>(W3 + (size_t)r * FO * FO, Wpk, wpkD);
    wmma_gemm_kernel<<<cdiv(tilesD, 4), dim3(128), 0, stream>>>(
        (const v8u*)Xpk, (const v8u*)Wpk, hA, FO / 32, tilesD);
    gather_kernel<<<gBlks, blk, 0, stream>>>(hA, eSrc[r], rowStart[r], doutC[r], acc, ND);
  }
  bias_act_kernel<<<cdiv(NF, 256), blk, 0, stream>>>(acc, b3, nullptr, (float*)d_out, NF, 0);
}
